// Aggregator_68796786147566
// MI455X (gfx1250) — matrix operand order repaired and checked by reading, not compiled, not run on hardware
//
#include <hip/hip_runtime.h>

// CDNA5 / gfx1250 fused GraphSAGE pooling aggregator.
// h = relu(nbr @ Wp + bp); pooled = max_k h; out = relu([src,pooled] @ W + b)
// bf16 WMMA (v_wmma_f32_16x16x32_bf16), f32 accumulate, wave32.
// Neighbor tiles staged via GLOBAL_LOAD_ASYNC_TO_LDS_B128 (ASYNCcnt-tracked DMA),
// double-buffered, one barrier per tile.

typedef __attribute__((ext_vector_type(16))) __bf16 v16bf;
typedef __attribute__((ext_vector_type(8)))  float  v8f;

#define DIMF     128
#define OUTF     128
#define KNBR     32
#define NPB      16            // nodes per block (grid = N/16 = 625, exact)
#define NTILES   (NPB * 2)     // 32 tiles of 16 neighbor rows per block
#define FPITCH   132           // f32 elems per LDS row (pad vs 64 banks)
#define PPITCH   132

// Build a 16-bit A/B fragment from f32: p[0..7] -> f[0..7], p[16..23] -> f[8..15]
__device__ __forceinline__ v16bf frag16_from_f32(const float* __restrict__ p) {
    v16bf a;
    #pragma unroll
    for (int j = 0; j < 8; ++j) { a[j] = (__bf16)p[j]; a[j + 8] = (__bf16)p[16 + j]; }
    return a;
}

__global__ __launch_bounds__(256)
void sage_pool_fused_kernel(const float* __restrict__ src,
                            const float* __restrict__ nbr,
                            const float* __restrict__ w_pool,
                            const float* __restrict__ bias_pool,
                            const float* __restrict__ w,
                            const float* __restrict__ bias,
                            float* __restrict__ out)
{
    __shared__ float aTile[2][16 * FPITCH];      // double-buffered 16x128 f32 tile
    __shared__ float pooled[NPB * PPITCH];       // 16 nodes x 128 pooled features

    const int tid   = threadIdx.x;
    const int wave  = tid >> 5;
    const int lane  = tid & 31;
    const int half  = lane >> 4;
    const int l16   = lane & 15;
    const int nBase = wave * 16;                 // this wave's 16 output columns
    const int node0 = blockIdx.x * NPB;

    // ---- GEMM1 B fragments: w_pool columns [nBase, nBase+16), kept in VGPRs ----
    v16bf b1[4];
    #pragma unroll
    for (int ks = 0; ks < 4; ++ks) {
        const int kb = ks * 32 + half * 8;
        #pragma unroll
        for (int j = 0; j < 16; ++j)
            b1[ks][j] = (__bf16)w_pool[(size_t)(kb + j + (j >= 8 ? 8 : 0)) * DIMF + nBase + l16];
    }
    const float bp = bias_pool[nBase + l16];

    // ---- async staging geometry: 256 threads x (2 x b128) cover 16x128 f32 ----
    const int srow = tid >> 4;
    const int scol = (tid & 15) * 8;             // 8 f32 = 32B per thread
    const float* gstage = nbr + ((size_t)node0 * KNBR + srow) * DIMF + scol;
    const unsigned ldsA[2] = {
        (unsigned)(size_t)&aTile[0][srow * FPITCH + scol],
        (unsigned)(size_t)&aTile[1][srow * FPITCH + scol] };

    // pipeline prologue: DMA tile 0 into buffer 0 (2 async b128 per thread)
    asm volatile("global_load_async_to_lds_b128 %0, %1, off\n\t"
                 "global_load_async_to_lds_b128 %0, %1, off offset:16"
                 :: "v"(ldsA[0]), "v"(gstage) : "memory");

    float maxv = -3.4e38f;
    int p = 0;
    for (int t = 0; t < NTILES; ++t) {
        // own async chunks for tile t have landed
        asm volatile("s_wait_asynccnt 0x0" ::: "memory");
        __syncthreads();                         // all waves' chunks landed; all
                                                 // reads of buffer p^1 (iter t-1) done
        // kick off DMA for tile t+1 into the other buffer (overlaps WMMA below)
        if (t + 1 < NTILES) {
            const float* gn = gstage + (size_t)(t + 1) * 16 * DIMF;
            asm volatile("global_load_async_to_lds_b128 %0, %1, off\n\t"
                         "global_load_async_to_lds_b128 %0, %1, off offset:16"
                         :: "v"(ldsA[p ^ 1]), "v"(gn) : "memory");
        }
        if (t + 4 < NTILES)                      // warm L2 ahead of the stream
            __builtin_prefetch(gstage + (size_t)(t + 4) * 16 * DIMF, 0, 3);

        // ---- consume tile t from buffer p: 4 x WMMA (K=128) ----
        const float* arow = &aTile[p][l16 * FPITCH];
        v8f acc = {};
        #pragma unroll
        for (int ks = 0; ks < 4; ++ks) {
            v16bf a = frag16_from_f32(arow + ks * 32 + half * 8);
            acc = __builtin_amdgcn_wmma_f32_16x16x32_bf16(
                      false, a, false, b1[ks], (short)0, acc, false, false);
        }
        // tile max over M (8 regs = half the rows, then cross-half shuffle)
        float tm = acc[0];
        #pragma unroll
        for (int i = 1; i < 8; ++i) tm = fmaxf(tm, acc[i]);
        tm   = fmaxf(tm, __shfl_xor(tm, 16, 32));
        maxv = fmaxf(maxv, tm);

        if (t & 1) {                             // both half-tiles of node t>>1 done
            // relu(max_k(x)+b) == max_k relu(x+b)
            if (half == 0)
                pooled[(t >> 1) * PPITCH + nBase + l16] = fmaxf(maxv + bp, 0.0f);
            maxv = -3.4e38f;
        }
        p ^= 1;
    }
    __syncthreads();

    // ---- GEMM2: out(16x128) = relu([src | pooled](16x256) @ W + b) ----
    v16bf b2[8];
    #pragma unroll
    for (int ks = 0; ks < 8; ++ks) {
        const int kb = ks * 32 + half * 8;
        #pragma unroll
        for (int j = 0; j < 16; ++j)
            b2[ks][j] = (__bf16)w[(size_t)(kb + j + (j >= 8 ? 8 : 0)) * OUTF + nBase + l16];
    }

    v8f acc = {};
    const float* srow2 = src + (size_t)(node0 + l16) * DIMF;
    #pragma unroll
    for (int ks = 0; ks < 4; ++ks) {             // K = 0..127 from src_features
        v16bf a = frag16_from_f32(srow2 + ks * 32 + half * 8);
        acc = __builtin_amdgcn_wmma_f32_16x16x32_bf16(
                  false, a, false, b2[ks], (short)0, acc, false, false);
    }
    const float* prow = &pooled[l16 * PPITCH];
    #pragma unroll
    for (int ks = 0; ks < 4; ++ks) {             // K = 128..255 from pooled (LDS)
        v16bf a = frag16_from_f32(prow + ks * 32 + half * 8);
        acc = __builtin_amdgcn_wmma_f32_16x16x32_bf16(
                  false, a, false, b2[ks + 4], (short)0, acc, false, false);
    }

    const float bb = bias[nBase + l16];
    #pragma unroll
    for (int i = 0; i < 8; ++i) {                // C layout: vgpr i -> M = i + half*8
        const int m = i + half * 8;
        out[(size_t)(node0 + m) * OUTF + nBase + l16] = fmaxf(acc[i] + bb, 0.0f);
    }
}

extern "C" void kernel_launch(void* const* d_in, const int* in_sizes, int n_in,
                              void* d_out, int out_size, void* d_ws, size_t ws_size,
                              hipStream_t stream) {
    const float* src       = (const float*)d_in[0];   // (10000,128)
    const float* nbr       = (const float*)d_in[1];   // (320000,128)
    const float* w_pool    = (const float*)d_in[2];   // (128,128)
    const float* bias_pool = (const float*)d_in[3];   // (128,)
    const float* w         = (const float*)d_in[4];   // (256,128)
    const float* bias      = (const float*)d_in[5];   // (128,)
    float*       out       = (float*)d_out;           // (10000,128)

    const int n_nodes = in_sizes[0] / DIMF;           // 10000, divisible by NPB
    dim3 grid((n_nodes + NPB - 1) / NPB), block(256);
    sage_pool_fused_kernel<<<grid, block, 0, stream>>>(src, nbr, w_pool, bias_pool,
                                                       w, bias, out);
}
